// SAE_84524956385365
// MI455X (gfx1250) — matrix operand order repaired and checked by reading, compiled, not run on hardware
//
#include <hip/hip_runtime.h>
#include <hip/hip_bf16.h>
#include <stdint.h>

// CDNA5 (gfx1250) WMMA types: bf16 A/B fragments (16 elems/lane), f32 C/D (8 elems/lane)
typedef __bf16 v16bf __attribute__((ext_vector_type(16)));
typedef float  v8f   __attribute__((ext_vector_type(8)));
// TDM descriptor groups (this toolchain: 6-arg builtin = g0 v4u, g1 v8i, g2 v4i, g3 v4i, v8i, cpol)
typedef unsigned int v4u __attribute__((ext_vector_type(4)));
typedef int          v8i __attribute__((ext_vector_type(8)));
typedef int          v4i __attribute__((ext_vector_type(4)));

#define TILE_K   32
#define BLK_M    128
#define BLK_N    128
#define ROW_DW   36                      // LDS row stride in DWORDs: 32 data + 4 TDM pad
#define TILE_DW  (BLK_M * ROW_DW)        // 4608 DW per staged tile
#define TILE_B   (TILE_DW * 4)           // 18432 bytes
#define SMEM_B   (4 * TILE_B)            // 2 buffers x (A,W) = 73728 bytes (dynamic LDS)

// ---- fp32 -> split bf16 (hi + lo) helpers (truncation; residual captured in lo) ----
__device__ __forceinline__ unsigned pack2_bf16_hi(float a, float b) {
  return (__float_as_uint(a) >> 16) | (__float_as_uint(b) & 0xFFFF0000u);
}
__device__ __forceinline__ float bf16_hi_part(float a) {
  return __uint_as_float(__float_as_uint(a) & 0xFFFF0000u);
}
__device__ __forceinline__ unsigned lds_addr_of(const void* p) {
  // generic pointer to LDS: low 32 bits are the LDS byte offset (ISA 10.2 aperture rules)
  return (unsigned)(uintptr_t)p;
}

// Issue one TDM 2D tile load: 128 rows x 32 fp32, row pitch K elems, +4DW LDS pad per row.
// D# per CDNA5 ISA ch.8: group0 = {ctl, lds_addr, gaddr_lo, gaddr_hi|type=2},
// group1 = {flags, dims/strides}, groups 2/3 unused (2D tile).
__device__ __forceinline__ void tdm_load_tile(const float* gptr, unsigned lds_byte, int K) {
  unsigned long long ga = (unsigned long long)(uintptr_t)gptr;
  v4u g0;
  g0[0] = 1u;                                        // count=1 (valid user descriptor)
  g0[1] = lds_byte;                                  // LDS destination (bytes)
  g0[2] = (unsigned)(ga & 0xFFFFFFFFull);            // global_addr[31:0]
  g0[3] = (unsigned)((ga >> 32) & 0x01FFFFFFull)     // global_addr[56:32]
          | (2u << 30);                              // type=2 ("image")
  v8i g1;
  g1[0] = 0x07120000;        // data_size=4B | pad_enable | pad_interval=32DW | pad_amount=4DW
  g1[1] = (int)(32u << 16);  // tensor_dim0 = 32 (tile-exact => no OOB)
  g1[2] = (int)(128u << 16); // tensor_dim0_hi=0 | tensor_dim1 = 128
  g1[3] = (int)(32u << 16);  // tensor_dim1_hi=0 | tile_dim0 = 32
  g1[4] = 128;               // tile_dim1 = 128 | tile_dim2 = 0
  g1[5] = K;                 // tensor_dim0_stride = K elements
  g1[6] = 0;
  g1[7] = 0;
  v4i z4 = {0, 0, 0, 0};
  v8i z8 = {0, 0, 0, 0, 0, 0, 0, 0};
  __builtin_amdgcn_tensor_load_to_lds(g0, g1, z4, z4, z8, 0);
}

// Convert 16 fp32 -> bf16 hi fragment + bf16 lo (residual) fragment
__device__ __forceinline__ void cvt16(const float* f, v16bf& hi, v16bf& lo) {
  union U { unsigned u[8]; v16bf v; } H, L;
#pragma unroll
  for (int i = 0; i < 8; ++i) {
    float a = f[2 * i], b = f[2 * i + 1];
    H.u[i] = pack2_bf16_hi(a, b);
    L.u[i] = pack2_bf16_hi(a - bf16_hi_part(a), b - bf16_hi_part(b));
  }
  hi = H.v;
  lo = L.v;
}

// A fragment (16x32): lane r holds K = half*8+0..7 (V0..3) and 16+half*8+0..7 (V4..7)
__device__ __forceinline__ void frag_a_f32(const float* tile, int row, int half,
                                           v16bf& hi, v16bf& lo) {
  union { float4 q[4]; float f[16]; } t;
  const float* p = tile + row * ROW_DW + half * 8;
  t.q[0] = *(const float4*)(p);
  t.q[1] = *(const float4*)(p + 4);
  t.q[2] = *(const float4*)(p + 16);
  t.q[3] = *(const float4*)(p + 20);
  cvt16(t.f, hi, lo);
}
// B fragment (32x16): lane n = column n, K = half*8+0..7 and 16+half*8+0..7
__device__ __forceinline__ void frag_b_f32(const float* tile, int col, int half,
                                           v16bf& hi, v16bf& lo) {
  union { float4 q[4]; float f[16]; } t;
  const float* p = tile + col * ROW_DW + half * 8;
  t.q[0] = *(const float4*)(p);
  t.q[1] = *(const float4*)(p + 4);
  t.q[2] = *(const float4*)(p + 16);
  t.q[3] = *(const float4*)(p + 20);
  cvt16(t.f, hi, lo);
}

// C[M,N] = act(A[M,K] * W[N,K]^T + bias[N]); fp32 in/out, split-bf16 WMMA accumulate.
// TDM double-buffered staging: wave 0 DMAs tile it+2 while all waves compute tile it+1.
template <bool RELU>
__global__ __launch_bounds__(256)
void sae_gemm(const float* __restrict__ A, const float* __restrict__ W,
              const float* __restrict__ bias, float* __restrict__ C,
              int N, int K) {
  extern __shared__ float sbuf[];   // [buf(2)][A,W][128][ROW_DW]

  const int tid  = threadIdx.x;
  const int lane = tid & 31;
  const int wv   = tid >> 5;   // 8 waves
  const int wm   = wv >> 1;    // 0..3 : wave M strip (32 rows)
  const int wn   = wv & 1;     // 0..1 : wave N strip (64 cols)
  const int lr   = lane & 15;
  const int half = lane >> 4;

  const int blockM = blockIdx.y * BLK_M;
  const int blockN = blockIdx.x * BLK_N;

  const float* Ab = A + (size_t)blockM * K;
  const float* Wb = W + (size_t)blockN * K;
  const unsigned lds0 = lds_addr_of(sbuf);
  const int steps = K / TILE_K;

  v8f acc[2][4] = {};

  // Prologue: fill both buffers asynchronously
  if (wv == 0) {
    tdm_load_tile(Ab,          lds0 + 0 * TILE_B, K);
    tdm_load_tile(Wb,          lds0 + 1 * TILE_B, K);
    tdm_load_tile(Ab + TILE_K, lds0 + 2 * TILE_B, K);
    tdm_load_tile(Wb + TILE_K, lds0 + 3 * TILE_B, K);
  }

  for (int it = 0; it < steps; ++it) {
    if (wv == 0) {
      if (it + 1 < steps) __builtin_amdgcn_s_wait_tensorcnt(2);  // oldest tile landed
      else                __builtin_amdgcn_s_wait_tensorcnt(0);  // drain at tail
    }
    __syncthreads();

    const float* tA = sbuf + (size_t)(it & 1) * (2 * TILE_DW);
    const float* tW = tA + TILE_DW;

    v16bf ahi[2], alo[2], bhi[4], blo[4];
#pragma unroll
    for (int mt = 0; mt < 2; ++mt)
      frag_a_f32(tA, wm * 32 + mt * 16 + lr, half, ahi[mt], alo[mt]);
#pragma unroll
    for (int nt = 0; nt < 4; ++nt)
      frag_b_f32(tW, wn * 64 + nt * 16 + lr, half, bhi[nt], blo[nt]);

#pragma unroll
    for (int mt = 0; mt < 2; ++mt)
#pragma unroll
      for (int nt = 0; nt < 4; ++nt) {
        acc[mt][nt] = __builtin_amdgcn_wmma_f32_16x16x32_bf16(
            false, ahi[mt], false, bhi[nt], (short)0, acc[mt][nt], false, false);
        acc[mt][nt] = __builtin_amdgcn_wmma_f32_16x16x32_bf16(
            false, ahi[mt], false, blo[nt], (short)0, acc[mt][nt], false, false);
        acc[mt][nt] = __builtin_amdgcn_wmma_f32_16x16x32_bf16(
            false, alo[mt], false, bhi[nt], (short)0, acc[mt][nt], false, false);
      }

    __syncthreads();  // all waves done reading buf (it&1) before TDM refills it
    if (wv == 0 && it + 2 < steps) {
      const int kk = (it + 2) * TILE_K;
      const unsigned b = (unsigned)(it & 1) * (2 * TILE_B);
      tdm_load_tile(Ab + kk, lds0 + b + 0 * TILE_B, K);
      tdm_load_tile(Wb + kk, lds0 + b + 1 * TILE_B, K);
    }
  }

  // Epilogue: C layout — lanes 0-15 = col n (M rows 0..7), lanes 16-31 same cols (M rows 8..15)
#pragma unroll
  for (int mt = 0; mt < 2; ++mt) {
#pragma unroll
    for (int nt = 0; nt < 4; ++nt) {
      int gcol = blockN + wn * 64 + nt * 16 + lr;
      int grow = blockM + wm * 32 + mt * 16 + half * 8;
      float bv = bias[gcol];
      float* cp = C + (size_t)grow * N + gcol;
#pragma unroll
      for (int v = 0; v < 8; ++v) {
        float r = acc[mt][nt][v] + bv;
        if (RELU) r = fmaxf(r, 0.0f);
        cp[(size_t)v * N] = r;
      }
    }
  }
}

// Per-row top-k mask via 4-pass radix select on fp32 bit patterns (ReLU => non-negative,
// so uint order == float order). One block per row; row stays L2-resident across passes.
__global__ __launch_bounds__(256)
void sae_topk(const float* __restrict__ h, float* __restrict__ ht, int L,
              const int* __restrict__ kp) {
  const int row = blockIdx.x;
  const int tid = threadIdx.x;
  const float* rp = h + (size_t)row * L;
  float* op = ht + (size_t)row * L;
  const int k = *kp;

  __shared__ unsigned hist[256];
  __shared__ unsigned sh_prefix;
  __shared__ int      sh_need;
  __shared__ unsigned sh_tie;
  if (tid == 0) { sh_prefix = 0u; sh_need = k; sh_tie = 0u; }
  __syncthreads();

#pragma unroll 1
  for (int pass = 0; pass < 4; ++pass) {
    const int shift = 24 - 8 * pass;
    const unsigned mhi = (pass == 0) ? 0u : (0xFFFFFFFFu << (shift + 8));
    hist[tid] = 0u;
    __syncthreads();
    const unsigned pfx = sh_prefix;
    for (int i = tid; i < L; i += 256) {
      unsigned u = __float_as_uint(rp[i]);
      if ((u & mhi) == pfx) atomicAdd(&hist[(u >> shift) & 0xFFu], 1u);
    }
    __syncthreads();
    if (tid == 0) {
      int need = sh_need, cum = 0, b = 255;
      for (; b > 0; --b) {
        int c = (int)hist[b];
        if (cum + c >= need) break;
        cum += c;
      }
      sh_prefix = pfx | ((unsigned)b << shift);
      sh_need = need - cum;  // ties at threshold still to take
    }
    __syncthreads();
  }

  const unsigned T = sh_prefix;          // exact bit pattern of k-th largest
  const unsigned ttake = (unsigned)sh_need;
  for (int i = tid; i < L; i += 256) {
    float v = rp[i];
    unsigned u = __float_as_uint(v);
    float o = 0.0f;
    if (u > T) o = v;
    else if (u == T && atomicAdd(&sh_tie, 1u) < ttake) o = v;
    op[i] = o;
  }
}

extern "C" void kernel_launch(void* const* d_in, const int* in_sizes, int n_in,
                              void* d_out, int out_size, void* d_ws, size_t ws_size,
                              hipStream_t stream) {
  const float* x     = (const float*)d_in[0];  // [4096, 2048]
  const float* W_enc = (const float*)d_in[1];  // [16384, 2048]
  const float* b_enc = (const float*)d_in[2];  // [16384]
  const float* W_dec = (const float*)d_in[3];  // [2048, 16384]
  const float* b_dec = (const float*)d_in[4];  // [2048]
  const int*   kp    = (const int*)d_in[5];    // scalar k

  const int B = 4096, D_IN = 2048, D_LAT = 16384;
  float* out    = (float*)d_out;
  float* x_hat  = out;                                       // [4096, 2048]
  float* h      = out + (size_t)B * D_IN;                    // [4096, 16384]
  float* h_topk = h + (size_t)B * D_LAT;                     // [4096, 16384]

  dim3 blk(256);
  // encode: h = relu(x @ W_enc^T + b_enc)   M=4096 N=16384 K=2048
  sae_gemm<true><<<dim3(D_LAT / BLK_N, B / BLK_M), blk, SMEM_B, stream>>>(
      x, W_enc, b_enc, h, D_LAT, D_IN);
  // top-k mask per row
  sae_topk<<<dim3(B), blk, 0, stream>>>(h, h_topk, D_LAT, kp);
  // decode: x_hat = h_topk @ W_dec^T + b_dec  M=4096 N=2048 K=16384
  sae_gemm<false><<<dim3(D_IN / BLK_N, B / BLK_M), blk, SMEM_B, stream>>>(
      h_topk, W_dec, b_dec, x_hat, D_IN, D_LAT);
}
